// SpatialBlock_70566312673727
// MI455X (gfx1250) — compile-verified, matrix operand order checked by reading, not run on hardware
//
#include <hip/hip_runtime.h>
#include <hip/hip_fp16.h>

typedef __attribute__((ext_vector_type(16))) _Float16 v16h;
typedef __attribute__((ext_vector_type(8)))  _Float16 v8h;
typedef __attribute__((ext_vector_type(8)))  float    v8f;

// ---------------------------------------------------------------------------
// A-fragment loader: 16 f32 from a row (k..k+7, k+16..k+23), scale, cvt f16.
// Matches the 16-bit 16x32 A layout (lane%16 = row, lane/16 = K half-group).
// ---------------------------------------------------------------------------
__device__ __forceinline__ v16h load_frag_g(const float* __restrict__ row, int kb, float scale) {
    float4 f0 = *reinterpret_cast<const float4*>(row + kb);
    float4 f1 = *reinterpret_cast<const float4*>(row + kb + 4);
    float4 g0 = *reinterpret_cast<const float4*>(row + kb + 16);
    float4 g1 = *reinterpret_cast<const float4*>(row + kb + 20);
    v16h r;
    r[0]  = (_Float16)(f0.x * scale); r[1]  = (_Float16)(f0.y * scale);
    r[2]  = (_Float16)(f0.z * scale); r[3]  = (_Float16)(f0.w * scale);
    r[4]  = (_Float16)(f1.x * scale); r[5]  = (_Float16)(f1.y * scale);
    r[6]  = (_Float16)(f1.z * scale); r[7]  = (_Float16)(f1.w * scale);
    r[8]  = (_Float16)(g0.x * scale); r[9]  = (_Float16)(g0.y * scale);
    r[10] = (_Float16)(g0.z * scale); r[11] = (_Float16)(g0.w * scale);
    r[12] = (_Float16)(g1.x * scale); r[13] = (_Float16)(g1.y * scale);
    r[14] = (_Float16)(g1.z * scale); r[15] = (_Float16)(g1.w * scale);
    return r;
}

// B-fragment from LDS-resident f16 weights (two ds_load_b128 per fragment).
__device__ __forceinline__ v16h load_frag_lds(const _Float16* p) {
    v8h lo = *reinterpret_cast<const v8h*>(p);
    v8h hi = *reinterpret_cast<const v8h*>(p + 16);
    v16h r;
#pragma unroll
    for (int i = 0; i < 8; ++i) { r[i] = lo[i]; r[8 + i] = hi[i]; }
    return r;
}

// ---------------------------------------------------------------------------
// out[r,:] = (scale1[r]*A1[r,:]) @ W1^T (+ A2[r,:] @ W2^T if DUAL) + bias
// W row-major [OUT, IN], staged once per block into LDS as f16.
// One wave computes a 16-row x OUT tile; 256 threads = 8 waves = 128 rows.
// nrows must be a multiple of 128; no divergence around WMMA (EXEC all-ones).
// ---------------------------------------------------------------------------
template<int IN, int OUT, bool DUAL, bool RELU>
__global__ void __launch_bounds__(256) gemm_wmma_kernel(
        const float* __restrict__ A1, const float* __restrict__ scale1,
        const float* __restrict__ W1,
        const float* __restrict__ A2, const float* __restrict__ W2,
        const float* __restrict__ bias, float* __restrict__ out)
{
    __shared__ __align__(16) _Float16 w1s[OUT * IN];
    __shared__ __align__(16) _Float16 w2s[DUAL ? OUT * IN : 8];

    for (int idx = threadIdx.x; idx < OUT * IN; idx += 256) {
        w1s[idx] = (_Float16)W1[idx];
        if (DUAL) w2s[idx] = (_Float16)W2[idx];
    }
    __syncthreads();

    const int wave = threadIdx.x >> 5;
    const int lane = threadIdx.x & 31;
    const int m    = lane & 15;      // A row in tile / B column
    const int kh   = lane >> 4;      // K half-group select
    const long long row0 = ((long long)blockIdx.x * 8 + wave) * 16;
    const long long r    = row0 + m;

    v8f acc[OUT / 16];
#pragma unroll
    for (int ot = 0; ot < OUT / 16; ++ot)
#pragma unroll
        for (int j = 0; j < 8; ++j) acc[ot][j] = 0.0f;

    {
        const float s = scale1 ? scale1[r] : 1.0f;
        const float* arow = A1 + r * IN;
#pragma unroll
        for (int ks = 0; ks < IN / 32; ++ks) {
            const int kb = ks * 32 + kh * 8;
            v16h a = load_frag_g(arow, kb, s);
#pragma unroll
            for (int ot = 0; ot < OUT / 16; ++ot) {
                v16h b = load_frag_lds(w1s + (ot * 16 + m) * IN + kb);
                acc[ot] = __builtin_amdgcn_wmma_f32_16x16x32_f16(
                    false, a, false, b, (short)0, acc[ot], false, false);
            }
        }
    }
    if (DUAL) {
        const float* arow = A2 + r * IN;
#pragma unroll
        for (int ks = 0; ks < IN / 32; ++ks) {
            const int kb = ks * 32 + kh * 8;
            v16h a = load_frag_g(arow, kb, 1.0f);
#pragma unroll
            for (int ot = 0; ot < OUT / 16; ++ot) {
                v16h b = load_frag_lds(w2s + (ot * 16 + m) * IN + kb);
                acc[ot] = __builtin_amdgcn_wmma_f32_16x16x32_f16(
                    false, a, false, b, (short)0, acc[ot], false, false);
            }
        }
    }

    // C/D layout: VGPR g holds row M = g + kh*8, column N = m.
#pragma unroll
    for (int ot = 0; ot < OUT / 16; ++ot) {
        const float bv = bias ? bias[ot * 16 + m] : 0.0f;
#pragma unroll
        for (int g = 0; g < 8; ++g) {
            float v = acc[ot][g] + bv;
            if (RELU) v = fmaxf(v, 0.0f);
            out[(row0 + g + kh * 8) * OUT + ot * 16 + m] = v;
        }
    }
}

// ---------------------------------------------------------------------------
// Elementwise / fill helpers
// ---------------------------------------------------------------------------
__global__ void fill_f32_kernel(float* __restrict__ p, float v, long long n) {
    long long i = (long long)blockIdx.x * blockDim.x + threadIdx.x;
    if (i < n) p[i] = v;
}
__global__ void fill_u32_kernel(unsigned* __restrict__ p, unsigned v, long long n) {
    long long i = (long long)blockIdx.x * blockDim.x + threadIdx.x;
    if (i < n) p[i] = v;
}
__global__ void bias_init_kernel(float* __restrict__ p, const float* __restrict__ b, long long n) {
    long long i = (long long)blockIdx.x * blockDim.x + threadIdx.x;
    if (i < n) p[i] = b[i & 63];
}
__global__ void relu_kernel(float* __restrict__ p, long long n) {
    long long i = (long long)blockIdx.x * blockDim.x + threadIdx.x;
    if (i < n) p[i] = fmaxf(p[i], 0.0f);
}
__global__ void invdeg_kernel(float* __restrict__ deg, int n) {
    int i = blockIdx.x * blockDim.x + threadIdx.x;
    if (i < n) deg[i] = 1.0f / fmaxf(deg[i], 1.0f);
}
__global__ void deg_kernel(const long long* __restrict__ dst, int E, float* __restrict__ deg) {
    int i = blockIdx.x * blockDim.x + threadIdx.x;
    if (i < E) atomicAdd(&deg[dst[i]], 1.0f);
}

// ---------------------------------------------------------------------------
// SAGE neighbor-sum scatter: 16 threads per edge, float4 gather + 4 f32
// atomics each; accumulator + activations are L2-resident (192 MB).
// ---------------------------------------------------------------------------
__global__ void sage_scatter_kernel(const float* __restrict__ h,
                                    const long long* __restrict__ src,
                                    const long long* __restrict__ dst,
                                    int E, float* __restrict__ agg) {
    long long t = (long long)blockIdx.x * blockDim.x + threadIdx.x;
    long long i = t >> 4;
    int f = (int)(t & 15) * 4;
    if (i >= E) return;
    long long s = src[i], d = dst[i];
    float4 v = *reinterpret_cast<const float4*>(h + s * 64 + f);
    float* o = agg + d * 64 + f;
    atomicAdd(o + 0, v.x); atomicAdd(o + 1, v.y);
    atomicAdd(o + 2, v.z); atomicAdd(o + 3, v.w);
}

// ---------------------------------------------------------------------------
// GAT pieces.  Ordered-uint key trick for float atomic max.
// ---------------------------------------------------------------------------
__device__ __forceinline__ unsigned fkey(float f) {
    unsigned u = __float_as_uint(f);
    return (u & 0x80000000u) ? ~u : (u | 0x80000000u);
}
__device__ __forceinline__ float fkey_inv(unsigned k) {
    return (k & 0x80000000u) ? __uint_as_float(k & 0x7fffffffu) : __uint_as_float(~k);
}

// one wave per node: asv[n] = hW[n,:].a_src ; adv[n] = hW[n,:].a_dst
__global__ void __launch_bounds__(256) gat_dots_kernel(
        const float* __restrict__ hW,
        const float* __restrict__ a_src, const float* __restrict__ a_dst,
        float* __restrict__ asv, float* __restrict__ adv) {
    const int wave = threadIdx.x >> 5, lane = threadIdx.x & 31;
    const long long node = (long long)blockIdx.x * 8 + wave;
    float h0 = hW[node * 64 + lane], h1 = hW[node * 64 + 32 + lane];
    float ps = h0 * a_src[lane] + h1 * a_src[32 + lane];
    float pd = h0 * a_dst[lane] + h1 * a_dst[32 + lane];
#pragma unroll
    for (int off = 16; off > 0; off >>= 1) {
        ps += __shfl_xor(ps, off, 32);
        pd += __shfl_xor(pd, off, 32);
    }
    if (lane == 0) { asv[node] = ps; adv[node] = pd; }
}

__global__ void gat_edge_max_kernel(const long long* __restrict__ src,
                                    const long long* __restrict__ dst,
                                    int E, int n,
                                    const float* __restrict__ asv,
                                    const float* __restrict__ adv,
                                    float* __restrict__ ebuf,
                                    unsigned* __restrict__ mkey) {
    long long i = (long long)blockIdx.x * blockDim.x + threadIdx.x;
    if (i >= (long long)E + n) return;
    long long s, d;
    if (i < E) { s = src[i]; d = dst[i]; } else { s = d = i - E; }
    float e = asv[s] + adv[d];
    e = (e > 0.0f) ? e : 0.2f * e;           // leaky_relu(0.2)
    ebuf[i] = e;
    atomicMax(&mkey[d], fkey(e));
}

__global__ void gat_edge_exp_kernel(const long long* __restrict__ dst,
                                    int E, int n,
                                    float* __restrict__ ebuf,
                                    const unsigned* __restrict__ mkey,
                                    float* __restrict__ denom) {
    long long i = (long long)blockIdx.x * blockDim.x + threadIdx.x;
    if (i >= (long long)E + n) return;
    long long d = (i < E) ? dst[i] : (i - E);
    float ee = __expf(ebuf[i] - fkey_inv(mkey[d]));
    ebuf[i] = ee;
    atomicAdd(&denom[d], ee);
}

__global__ void gat_edge_alpha_kernel(const long long* __restrict__ dst,
                                      int E, int n,
                                      float* __restrict__ ebuf,
                                      const float* __restrict__ denom) {
    long long i = (long long)blockIdx.x * blockDim.x + threadIdx.x;
    if (i >= (long long)E + n) return;
    long long d = (i < E) ? dst[i] : (i - E);
    ebuf[i] = ebuf[i] / fmaxf(denom[d], 1e-16f);
}

// 16 threads per edge, float4 gather scaled by alpha, 4 atomics each.
__global__ void gat_scatter_kernel(const long long* __restrict__ src,
                                   const long long* __restrict__ dst,
                                   int E, int n,
                                   const float* __restrict__ hW,
                                   const float* __restrict__ alpha,
                                   float* __restrict__ out) {
    long long t = (long long)blockIdx.x * blockDim.x + threadIdx.x;
    long long i = t >> 4;
    int f = (int)(t & 15) * 4;
    if (i >= (long long)E + n) return;
    long long s, d;
    if (i < E) { s = src[i]; d = dst[i]; } else { s = d = i - E; }
    float a = alpha[i];
    float4 v = *reinterpret_cast<const float4*>(hW + s * 64 + f);
    float* o = out + d * 64 + f;
    atomicAdd(o + 0, a * v.x); atomicAdd(o + 1, a * v.y);
    atomicAdd(o + 2, a * v.z); atomicAdd(o + 3, a * v.w);
}

// ---------------------------------------------------------------------------
// Orchestration
// ---------------------------------------------------------------------------
extern "C" void kernel_launch(void* const* d_in, const int* in_sizes, int n_in,
                              void* d_out, int out_size, void* d_ws, size_t ws_size,
                              hipStream_t stream) {
    const float*     x       = (const float*)d_in[0];
    const long long* ei      = (const long long*)d_in[1];   // int64 edge_index [2,E]
    const float*     sage_Wl = (const float*)d_in[2];
    const float*     sage_Wr = (const float*)d_in[3];
    const float*     sage_b  = (const float*)d_in[4];
    const float*     gat_W   = (const float*)d_in[5];
    const float*     gat_as  = (const float*)d_in[6];
    const float*     gat_ad  = (const float*)d_in[7];
    const float*     gat_b   = (const float*)d_in[8];
    const float*     p1_W    = (const float*)d_in[9];
    const float*     p1_b    = (const float*)d_in[10];
    const float*     p2_W    = (const float*)d_in[11];
    const float*     p2_b    = (const float*)d_in[12];
    const float*     p3_W    = (const float*)d_in[13];
    const float*     p3_b    = (const float*)d_in[14];

    const int n = in_sizes[0] / 64;
    const int E = in_sizes[1] / 2;
    const long long* src = ei;
    const long long* dst = ei + E;

    // workspace layout (~213 MB): three N*64 ping-pong buffers + small strips
    float*    bufA   = (float*)d_ws;
    float*    bufB   = bufA + (size_t)n * 64;
    float*    bufC   = bufB + (size_t)n * 64;
    float*    invdeg = bufC + (size_t)n * 64;
    float*    asv    = invdeg + n;
    float*    adv    = asv + n;
    float*    denom  = adv + n;
    unsigned* mkey   = (unsigned*)(denom + n);
    float*    ebuf   = (float*)(mkey + n);

    const dim3 B(256);
    auto cdiv = [](long long a, long long b) { return (unsigned)((a + b - 1) / b); };
    const long long NF = (long long)n * 64;
    const int gemm_grid = n / 128;

    // --- degree -> inverse degree (shared by all SAGE layers) ---
    fill_f32_kernel<<<cdiv(n, 256), B, 0, stream>>>(invdeg, 0.0f, n);
    deg_kernel<<<cdiv(E, 256), B, 0, stream>>>(dst, E, invdeg);
    invdeg_kernel<<<cdiv(n, 256), B, 0, stream>>>(invdeg, n);

    // --- 4 x GraphSAGE ---
    const float* h = x;
    for (int i = 0; i < 4; ++i) {
        fill_f32_kernel<<<cdiv(NF, 256), B, 0, stream>>>(bufC, 0.0f, NF);
        sage_scatter_kernel<<<cdiv((long long)E * 16, 256), B, 0, stream>>>(h, src, dst, E, bufC);
        float* out = (i & 1) ? bufB : bufA;
        if (i < 3)
            gemm_wmma_kernel<64, 64, true, true><<<gemm_grid, B, 0, stream>>>(
                bufC, invdeg, sage_Wl + i * 4096, h, sage_Wr + i * 4096, sage_b + i * 64, out);
        else
            gemm_wmma_kernel<64, 64, true, false><<<gemm_grid, B, 0, stream>>>(
                bufC, invdeg, sage_Wl + i * 4096, h, sage_Wr + i * 4096, sage_b + i * 64, out);
        h = out;
    }

    // --- 3 x GAT (with self loops) ---
    const long long Ep = (long long)E + n;
    for (int i = 0; i < 3; ++i) {
        float* hW;
        float* out;
        if (h == bufB)      { hW = bufA; out = bufC; }
        else if (h == bufC) { hW = bufA; out = bufB; }
        else                { hW = bufB; out = bufC; }

        // hW = h @ W^T (no bias)
        gemm_wmma_kernel<64, 64, false, false><<<gemm_grid, B, 0, stream>>>(
            h, nullptr, gat_W + i * 4096, nullptr, nullptr, nullptr, hW);

        gat_dots_kernel<<<n / 8, B, 0, stream>>>(hW, gat_as + i * 64, gat_ad + i * 64, asv, adv);

        fill_u32_kernel<<<cdiv(n, 256), B, 0, stream>>>(mkey, 0u, n);
        fill_f32_kernel<<<cdiv(n, 256), B, 0, stream>>>(denom, 0.0f, n);

        gat_edge_max_kernel<<<cdiv(Ep, 256), B, 0, stream>>>(src, dst, E, n, asv, adv, ebuf, mkey);
        gat_edge_exp_kernel<<<cdiv(Ep, 256), B, 0, stream>>>(dst, E, n, ebuf, mkey, denom);
        gat_edge_alpha_kernel<<<cdiv(Ep, 256), B, 0, stream>>>(dst, E, n, ebuf, denom);

        // fold bias into the scatter target, then accumulate alpha * hW[src]
        bias_init_kernel<<<cdiv(NF, 256), B, 0, stream>>>(out, gat_b + i * 64, NF);
        gat_scatter_kernel<<<cdiv(Ep * 16, 256), B, 0, stream>>>(src, dst, E, n, hW, ebuf, out);
        if (i < 2) relu_kernel<<<cdiv(NF, 256), B, 0, stream>>>(out, NF);
        h = out;
    }

    // --- projection head (64->64 relu, 64->32 relu, 32->16) ---
    gemm_wmma_kernel<64, 64, false, true><<<gemm_grid, B, 0, stream>>>(
        h, nullptr, p1_W, nullptr, nullptr, p1_b, bufA);
    gemm_wmma_kernel<64, 32, false, true><<<gemm_grid, B, 0, stream>>>(
        bufA, nullptr, p2_W, nullptr, nullptr, p2_b, bufB);
    gemm_wmma_kernel<32, 16, false, false><<<gemm_grid, B, 0, stream>>>(
        bufB, nullptr, p3_W, nullptr, nullptr, p3_b, (float*)d_out);
}
